// VQVAE_43679817400604
// MI455X (gfx1250) — matrix operand order repaired and checked by reading, not compiled, not run on hardware
//
#include <hip/hip_runtime.h>
#include <hip/hip_bf16.h>
#include <math.h>
#include <stdint.h>

// ---------------------------------------------------------------------------
// VQ-VAE bottleneck for MI455X (gfx1250, wave32, WMMA bf16 16x16x32 + TDM)
// ---------------------------------------------------------------------------

typedef __attribute__((ext_vector_type(16))) __bf16 v16bf;
typedef __attribute__((ext_vector_type(8)))  float  v8f;
typedef __attribute__((ext_vector_type(4)))  unsigned u32x4;
typedef __attribute__((ext_vector_type(8)))  int      i32x8;
typedef __attribute__((ext_vector_type(4)))  int      i32x4;

#define NDIM  16
#define WDIM  512
#define TDIM  8192
#define KBINS 2048
#define NT    (NDIM * TDIM)          // 131072 rows

// workspace byte offsets
#define WS_SCAL  0                               // 3 floats: Sx, Sxx, MinSum
#define WS_K2    64                              // 2048 floats
#define WS_KBF   (64 + KBINS * 4)                // 2048*512 bf16
#define WS_CODES (WS_KBF + KBINS * WDIM * 2)     // NT ints

#define ROWP 260   // padded LDS row in dwords (520 bf16) -> conflict-free strides

#if __has_builtin(__builtin_amdgcn_tensor_load_to_lds) && \
    __has_builtin(__builtin_amdgcn_s_wait_tensorcnt)
#define USE_TDM 1
#else
#define USE_TDM 0
#endif

__device__ __forceinline__ unsigned f2bf1(float f) {
  unsigned u = __float_as_uint(f);
  return (u + 0x7FFFu + ((u >> 16) & 1u)) >> 16;   // round-to-nearest-even
}

// ---------------------------------------------------------------------------
// Kernel: zero the three accumulator scalars
// ---------------------------------------------------------------------------
__global__ void vq_init(float* scal) {
  if (threadIdx.x < 3) scal[threadIdx.x] = 0.f;
}

// ---------------------------------------------------------------------------
// Kernel: codebook prep  k (fp32) -> k_bf16, k2[c] = sum_w k[c][w]^2
// grid: KBINS blocks x 128 threads
// ---------------------------------------------------------------------------
__global__ void vq_prep_k(const float* __restrict__ k,
                          unsigned short* __restrict__ kbf,
                          float* __restrict__ k2) {
  __shared__ float wsum[4];
  const int c = blockIdx.x, tid = threadIdx.x;
  const float4 f = ((const float4*)(k + (size_t)c * WDIM))[tid];
  float s = f.x * f.x + f.y * f.y + f.z * f.z + f.w * f.w;
  uint2 p;
  p.x = f2bf1(f.x) | (f2bf1(f.y) << 16);
  p.y = f2bf1(f.z) | (f2bf1(f.w) << 16);
  ((uint2*)kbf)[(size_t)c * 128 + tid] = p;
#pragma unroll
  for (int off = 16; off; off >>= 1) s += __shfl_xor(s, off, 32);
  if ((tid & 31) == 0) wsum[tid >> 5] = s;
  __syncthreads();
  if (tid == 0) k2[c] = wsum[0] + wsum[1] + wsum[2] + wsum[3];
}

// ---------------------------------------------------------------------------
// Main kernel: distances + argmin via bf16 WMMA, k chunks staged by the TDM.
// grid: 1024 blocks (16 n * 64 t-tiles) x 256 threads (8 waves).
// Block: M=128 rows (t0..t0+127 of one n) x all 2048 codes (16 chunks of 128).
// Wave (waveM in 0..3, waveN in 0..1): M=32 x N=64 register tile
//   -> per K-step: 2 A frags + 4 B frags -> 8 v_wmma_f32_16x16x32_bf16.
// ---------------------------------------------------------------------------
__global__ __launch_bounds__(256, 1)
void vq_main(const float* __restrict__ x, const float* __restrict__ k2g,
             const unsigned short* __restrict__ kbf,
             float* __restrict__ out_xl, int* __restrict__ codes,
             float* __restrict__ scal) {
  extern __shared__ __align__(16) char smem[];
  unsigned* aT  = (unsigned*)smem;           // 128 * ROWP dwords (x tile, bf16)
  unsigned* bT  = aT + 128 * ROWP;           // 128 * ROWP dwords (k chunk, bf16)
  float*    x2b = (float*)(bT + 128 * ROWP); // 256 floats (row |x|^2)
  float*    minb = x2b + 256;                // 256 floats (merge)
  int*      argb = (int*)(minb + 256);       // 256 ints  (merge)

  const int tid  = threadIdx.x;
  const int lane = tid & 31;
  const int wv   = tid >> 5;
  const int hi   = lane >> 4;     // half-wave select
  const int l16  = lane & 15;
  const int waveM = wv & 3, waveN = wv >> 2;
  const int n  = blockIdx.x >> 6;
  const int t0 = (blockIdx.x & 63) << 7;

  // ---- Phase 1: stage x tile (128 t x 512 w) as bf16 into LDS; stats ----
  const float* xb = x + (size_t)n * WDIM * TDIM + t0;
  const int tloc = tid & 127;
  const int wb   = tid >> 7;
  float sx = 0.f, sq = 0.f;
#pragma unroll 4
  for (int it = 0; it < 128; ++it) {
    const int wp = wb + 2 * it;     // w-pair index 0..255
    const int w0 = wp * 2;
    const float a = xb[(size_t)w0 * TDIM + tloc];
    const float b = xb[(size_t)(w0 + 1) * TDIM + tloc];
    sx += a + b;
    sq += a * a + b * b;
    aT[tloc * ROWP + wp] = f2bf1(a) | (f2bf1(b) << 16);
  }
  x2b[tid] = sq;                    // per-(row,half) partial |x|^2
  float rsx = sx, rsq = sq;
#pragma unroll
  for (int off = 16; off; off >>= 1) {
    rsx += __shfl_xor(rsx, off, 32);
    rsq += __shfl_xor(rsq, off, 32);
  }
  if (lane == 0) { atomicAdd(&scal[0], rsx); atomicAdd(&scal[1], rsq); }
  __syncthreads();
  if (tid < 128) x2b[tid] = x2b[tid] + x2b[tid + 128];  // full row |x|^2

  // ---- Phase 2: loop code chunks, WMMA, running argmin of (k2 - 2*dot) ----
  const uint4* kbf4 = (const uint4*)kbf;
  float minv[2][8];
  int   mina[2][8];
#pragma unroll
  for (int s = 0; s < 2; ++s)
#pragma unroll
    for (int v = 0; v < 8; ++v) { minv[s][v] = 3.4e38f; mina[s][v] = 0; }

  const int rowA0   = waveM * 32 + l16;   // + s*16 per strip
  const int colBase = waveN * 64 + l16;   // + t*16 per N-tile

  union FB { v16bf v; uint4 q[2]; };

  for (int ch = 0; ch < 16; ++ch) {
    const int c0 = ch * 128;
    __syncthreads();                      // prior chunk compute done (WAR on bT)

#if USE_TDM
    // ---- Tensor Data Mover: DMA 128 codes x 512 bf16 into padded LDS ----
    if (wv == 0) {
      const unsigned long long ga =
          (unsigned long long)(uintptr_t)kbf + (unsigned long long)c0 * WDIM * 2ull;
      const unsigned ldsb = (unsigned)(uintptr_t)bT;  // LDS byte offset
      u32x4 g0;
      g0[0] = 1u;                                     // count=1, 2D, no gather
      g0[1] = ldsb;                                   // lds_addr
      g0[2] = (unsigned)(ga & 0xFFFFFFFFu);           // global_addr[31:0]
      g0[3] = (unsigned)((ga >> 32) & 0x01FFFFFFu)    // global_addr[56:32]
              | (2u << 30);                           // type = 2 ("image")
      i32x8 g1;
      g1[0] = (int)((1u << 16)    // data_size = 2B
                  | (1u << 20)    // pad_enable
                  | (7u << 22)    // pad_interval: every 256 dwords (one row)
                  | (3u << 25));  // pad_amount: 4 dwords (8 bf16) -> ROWP=260
      g1[1] = (int)(512u << 16);                  // tensor_dim0 = 512
      g1[2] = (int)((2048u & 0xFFFFu) << 16);     // tensor_dim1 = 2048
      g1[3] = (int)(512u << 16);                  // tile_dim0 = 512
      g1[4] = (int)128u;                          // tile_dim1 = 128, tile_dim2 = 0
      g1[5] = (int)512;                           // tensor_dim0_stride = 512
      g1[6] = 0;
      g1[7] = 0;
      const i32x4 z4 = {0, 0, 0, 0};
#if __clang_major__ >= 23
      const i32x8 z8 = {0, 0, 0, 0, 0, 0, 0, 0};
      __builtin_amdgcn_tensor_load_to_lds(g0, g1, z4, z4, z8, 0);
#else
      __builtin_amdgcn_tensor_load_to_lds(g0, g1, z4, z4, 0);
#endif
      __builtin_amdgcn_s_wait_tensorcnt(0);
    }
#else
    // ---- fallback: manual global b128 -> ds b128 staging ----
#pragma unroll 4
    for (int j = tid; j < 128 * 64; j += 256) {
      const int c = j >> 6, q = j & 63;
      uint4 vv = kbf4[(size_t)(c0 + c) * 64 + q];
      *(uint4*)(bT + c * ROWP + q * 4) = vv;
    }
#endif
    if (ch < 15 && tid < 128)             // warm cache for next chunk
      __builtin_prefetch(kbf4 + (size_t)(c0 + 128 + tid) * 64, 0, 1);
    __syncthreads();

    v8f acc[2][4];
    const v8f z = {0.f, 0.f, 0.f, 0.f, 0.f, 0.f, 0.f, 0.f};
#pragma unroll
    for (int s = 0; s < 2; ++s)
#pragma unroll
      for (int t = 0; t < 4; ++t) acc[s][t] = z;

    for (int kk = 0; kk < 16; ++kk) {     // K = 512 in steps of 32
      FB fa[2], fb[4];
      // A frag: 16 consecutive K at w=kk*32+16*hi (matches B)
#pragma unroll
      for (int s = 0; s < 2; ++s) {
        const unsigned* p = aT + (rowA0 + s * 16) * ROWP + kk * 16 + hi * 8;
        fa[s].q[0] = *(const uint4*)p;
        fa[s].q[1] = *(const uint4*)(p + 4);
      }
      // B frag (same k order as A): 16 consecutive K at w=kk*32+16*hi
#pragma unroll
      for (int t = 0; t < 4; ++t) {
        const unsigned* p = bT + (colBase + t * 16) * ROWP + kk * 16 + hi * 8;
        fb[t].q[0] = *(const uint4*)p;
        fb[t].q[1] = *(const uint4*)(p + 4);
      }
#pragma unroll
      for (int s = 0; s < 2; ++s)
#pragma unroll
        for (int t = 0; t < 4; ++t)
          acc[s][t] = __builtin_amdgcn_wmma_f32_16x16x32_bf16(
              false, fa[s].v, false, fb[t].v, (short)0, acc[s][t], false, false);
    }

    // epilogue: m = k2[c] - 2*dot  (x^2 is row-constant; added only for fit)
#pragma unroll
    for (int t = 0; t < 4; ++t) {
      const int c = c0 + colBase + t * 16;
      const float kc = k2g[c];
#pragma unroll
      for (int s = 0; s < 2; ++s)
#pragma unroll
        for (int v = 0; v < 8; ++v) {
          const float m = __builtin_fmaf(-2.f, acc[s][t][v], kc);
          if (m < minv[s][v]) { minv[s][v] = m; mina[s][v] = c; }
        }
    }
  }

  // ---- cross-lane argmin within each 16-lane half (wave32 shuffles) ----
#pragma unroll
  for (int s = 0; s < 2; ++s)
#pragma unroll
    for (int v = 0; v < 8; ++v) {
      float m = minv[s][v];
      int   a = mina[s][v];
#pragma unroll
      for (int off = 1; off < 16; off <<= 1) {
        const float om = __shfl_xor(m, off, 32);
        const int   oa = __shfl_xor(a, off, 32);
        if (om < m || (om == m && oa < a)) { m = om; a = oa; }
      }
      minv[s][v] = m; mina[s][v] = a;
    }
  if (l16 == 0) {
#pragma unroll
    for (int s = 0; s < 2; ++s)
#pragma unroll
      for (int v = 0; v < 8; ++v) {
        const int row = waveM * 32 + s * 16 + hi * 8 + v;
        minb[waveN * 128 + row] = minv[s][v];
        argb[waveN * 128 + row] = mina[s][v];
      }
  }
  __syncthreads();

  // ---- merge the two N-halves, emit codes + fit contribution ----
  if (tid < 128) {
    const int row = tid;
    float m0 = minb[row];             int a0 = argb[row];
    const float m1 = minb[128 + row]; const int a1 = argb[128 + row];
    if (m1 < m0 || (m1 == m0 && a1 < a0)) { m0 = m1; a0 = a1; }
    const size_t gi = (size_t)n * TDIM + t0 + row;
    out_xl[gi] = (float)a0;
    codes[gi]  = a0;
    float fitc = m0 + x2b[row];       // full squared distance
#pragma unroll
    for (int off = 16; off; off >>= 1) fitc += __shfl_xor(fitc, off, 32);
    if (lane == 0) atomicAdd(&scal[2], fitc);
  }
}

// ---------------------------------------------------------------------------
// Gather: x_q[n][w][t] = k[code[n,t]][w]   (straight-through forward value)
// ---------------------------------------------------------------------------
__global__ void vq_gather(const int* __restrict__ codes,
                          const float* __restrict__ k,
                          float* __restrict__ xq) {
  const int idx = blockIdx.x * 256 + threadIdx.x;  // row in [0, NT)
  const int n = idx >> 13;
  const int t = idx & (TDIM - 1);
  const int c = codes[idx];
  const float* kr = k + (size_t)c * WDIM;
  float* o = xq + (size_t)n * WDIM * TDIM + t;
#pragma unroll 8
  for (int w = 0; w < WDIM; ++w) o[(size_t)w * TDIM] = kr[w];
}

// ---------------------------------------------------------------------------
// Finalize scalars
// ---------------------------------------------------------------------------
__global__ void vq_fin(const float* __restrict__ scal, float* __restrict__ out_sc) {
  if (threadIdx.x != 0) return;
  const double nel = (double)NT * (double)WDIM;
  const double sxv = scal[0], sxx = scal[1], ms = scal[2];
  const double mean = sxv / nel;
  double vs = sxx - nel * mean * mean;
  if (vs < 0) vs = 0;
  out_sc[0] = (float)(ms / nel);        // commit_loss
  out_sc[1] = (float)(ms / (double)NT); // fit
  out_sc[2] = (float)sqrt(vs / nel);    // prenorm
}

// ---------------------------------------------------------------------------
extern "C" void kernel_launch(void* const* d_in, const int* in_sizes, int n_in,
                              void* d_out, int out_size, void* d_ws, size_t ws_size,
                              hipStream_t stream) {
  const float* x = (const float*)d_in[0];   // [16, 512, 8192] fp32
  const float* k = (const float*)d_in[1];   // [2048, 512]     fp32
  float* out = (float*)d_out;
  char*  ws  = (char*)d_ws;

  float*          scal  = (float*)(ws + WS_SCAL);
  float*          k2    = (float*)(ws + WS_K2);
  unsigned short* kbf   = (unsigned short*)(ws + WS_KBF);
  int*            codes = (int*)(ws + WS_CODES);

  float* out_xl = out;                                    // NT floats
  float* out_xq = out + NT;                               // N*W*T floats
  float* out_sc = out + NT + (size_t)NDIM * WDIM * TDIM;  // 3 floats

  vq_init<<<1, 32, 0, stream>>>(scal);
  vq_prep_k<<<KBINS, 128, 0, stream>>>(k, kbf, k2);
  const size_t shmem = (size_t)(2 * 128 * ROWP + 256 + 256 + 256) * 4;  // ~263 KB
  vq_main<<<NDIM * (TDIM / 128), 256, shmem, stream>>>(x, k2, kbf, out_xl, codes, scal);
  vq_gather<<<NT / 256, 256, 0, stream>>>(codes, k, out_xq);
  vq_fin<<<1, 32, 0, stream>>>(scal, out_sc);
}
